// MaskedMultiHeadAttention_68891275428471
// MI455X (gfx1250) — matrix operand order repaired and checked by reading, not compiled, not run on hardware
//
#include <hip/hip_runtime.h>

typedef __bf16 bf16_t;
typedef __attribute__((ext_vector_type(16))) __bf16 v16bf;
typedef __attribute__((ext_vector_type(8)))  __bf16 v8bf;
typedef __attribute__((ext_vector_type(4)))  __bf16 v4bf;
typedef __attribute__((ext_vector_type(8)))  float  v8f;
typedef __attribute__((ext_vector_type(4)))  float  v4f;

#define MHA_B 4
#define MHA_L 2048
#define MHA_D 1024

#define BM 128
#define BN 128
#define BK 32
#define LDSK (BK + 8)                       // 80 B rows, 16 B aligned
#define PLANE_BYTES (BM * LDSK * 2)         // 10240: A->B plane stride
#define BUF_BYTES   (2 * PLANE_BYTES)       // 20480: double-buffer stride

union FragBf { v16bf v; v8bf h[2]; };

// LDS byte address of a __shared__ object (generic -> AS3 -> int)
static __device__ __forceinline__ unsigned lds_addr(const void* p) {
  return (unsigned)(unsigned long long)(__attribute__((address_space(3))) const void*)p;
}

template <int N>
static __device__ __forceinline__ void wait_asynccnt() {
#if __has_builtin(__builtin_amdgcn_s_wait_asynccnt)
  __builtin_amdgcn_s_wait_asynccnt((short)N);
#else
  if constexpr (N == 0)
    asm volatile("s_wait_asynccnt 0x0" ::: "memory");
  else
    asm volatile("s_wait_asynccnt 0x8" ::: "memory");
#endif
}

static __device__ __forceinline__ void async_cp_b128(unsigned ldsOff,
                                                     unsigned long long gaddr) {
  asm volatile("global_load_async_to_lds_b128 %0, %1, off"
               :: "v"(ldsOff), "v"(gaddr) : "memory");
}

// ---------------------------------------------------------------- f32 -> bf16
__global__ __launch_bounds__(256) void cvt_f32_bf16(const float* __restrict__ in,
                                                    bf16_t* __restrict__ out,
                                                    long n) {
  long i = ((long)blockIdx.x * 256 + threadIdx.x) * 4;
  if (i + 3 < n) {
    v4f x = *(const v4f*)(in + i);
    v4bf r;
    r.x = (bf16_t)x.x; r.y = (bf16_t)x.y; r.z = (bf16_t)x.z; r.w = (bf16_t)x.w;
    *(v4bf*)(out + i) = r;
  }
}

// ------------------------------------------------------- bf16 NT GEMM (WMMA)
// C[m,n] = alpha * sum_k A[m,k] * Bt[n,k]; lda = ldb = K.
// STORE_MODE: 0 = f32 row-major, 1 = bf16 row-major,
//             2 = bf16 per-batch transposed: out[(gm/Lrows)*N + gn][gm%Lrows]
template <int STORE_MODE>
__global__ __launch_bounds__(128) void gemm_bf16_nt(
    const bf16_t* __restrict__ A, const bf16_t* __restrict__ Bt,
    void* __restrict__ Cout, int M, int N, int K,
    long strideA, long strideB, long strideC, float alpha, int Lrows) {
  // [buf][A|B][row][k] double-buffered tiles, filled by async DMA
  __shared__ __align__(16) bf16_t smem[2][2][BM][LDSK];

  const int tid  = threadIdx.x;
  const int lane = tid & 31;
  const int wave = tid >> 5;
  const int z    = blockIdx.z;

  const bf16_t* a = A + (long)z * strideA;
  const bf16_t* b = Bt + (long)z * strideB;
  const long cbase = (long)z * strideC;

  const int tileM = blockIdx.y * BM;
  const int tileN = blockIdx.x * BN;
  const int waveM = (wave >> 1) * 64;
  const int waveN = (wave & 1) * 64;

  v8f acc[4][4];
#pragma unroll
  for (int i = 0; i < 4; ++i)
#pragma unroll
    for (int j = 0; j < 4; ++j) acc[i][j] = v8f{};

  // ISA 7.12.2 fragment maps (wave32):
  //  A 16x32 bf16: m = lane&15; half = lane>>4; elems -> K = half*8(+16) + j
  //  B 32x16 bf16: n = lane&15; elems -> K = half*8(+16) + j
  //  C/D f32: elem e -> m = half*8 + e, n = lane&15
  const int fm   = lane & 15;
  const int half = lane >> 4;
  const int aks  = half * 8;
  const int bks  = half * 8;

  // async staging: 4 threads cover one 32-elem row (16 B each)
  const int ldChunk = (tid & 3) * 8;
  const int ldRow0  = tid >> 2;  // 0..31, rows ldRow0 + 32*i

  const unsigned ldsBase = lds_addr(&smem[0][0][ldRow0][ldChunk]);
  unsigned long long ga[4], gb[4];
#pragma unroll
  for (int i = 0; i < 4; ++i) {
    ga[i] = (unsigned long long)(uintptr_t)(a + (long)(tileM + ldRow0 + i * 32) * K + ldChunk);
    gb[i] = (unsigned long long)(uintptr_t)(b + (long)(tileN + ldRow0 + i * 32) * K + ldChunk);
  }

  auto issue = [&](int bf) {  // 8 async b128 DMAs, then advance one K-slab
    const unsigned lb = ldsBase + (unsigned)bf * BUF_BYTES;
#pragma unroll
    for (int i = 0; i < 4; ++i) {
      async_cp_b128(lb + i * (32 * LDSK * 2), ga[i]);
      async_cp_b128(lb + PLANE_BYTES + i * (32 * LDSK * 2), gb[i]);
      ga[i] += BK * 2;
      gb[i] += BK * 2;
    }
  };

  issue(0);
  int buf = 0;
  for (int k0 = 0; k0 < K; k0 += BK) {
    if (k0 + BK < K) {
      issue(buf ^ 1);
      wait_asynccnt<8>();   // oldest group (this buf) has landed
    } else {
      wait_asynccnt<0>();
    }
    __syncthreads();

    const bf16_t (*As)[LDSK] = smem[buf][0];
    const bf16_t (*Bs)[LDSK] = smem[buf][1];

    v16bf af[4], bfv[4];
#pragma unroll
    for (int i = 0; i < 4; ++i) {
      FragBf fa;
      const int m = waveM + i * 16 + fm;
      fa.h[0] = *(const v8bf*)&As[m][aks];
      fa.h[1] = *(const v8bf*)&As[m][aks + 16];
      af[i] = fa.v;
      FragBf fb;
      const int n = waveN + i * 16 + fm;
      fb.h[0] = *(const v8bf*)&Bs[n][bks];
      fb.h[1] = *(const v8bf*)&Bs[n][bks + 16];
      bfv[i] = fb.v;
    }
#pragma unroll
    for (int i = 0; i < 4; ++i)
#pragma unroll
      for (int j = 0; j < 4; ++j)
        acc[i][j] = __builtin_amdgcn_wmma_f32_16x16x32_bf16(
            false, af[i], false, bfv[j], (short)0, acc[i][j], false, false);

    __syncthreads();  // protects buf from next iteration's DMA
    buf ^= 1;
  }

  const int cn = lane & 15;
#pragma unroll
  for (int i = 0; i < 4; ++i) {
#pragma unroll
    for (int j = 0; j < 4; ++j) {
#pragma unroll
      for (int e = 0; e < 8; ++e) {
        const float val = acc[i][j][e] * alpha;
        const int gm = tileM + waveM + i * 16 + half * 8 + e;
        const int gn = tileN + waveN + j * 16 + cn;
        if (STORE_MODE == 0) {
          ((float*)Cout)[cbase + (long)gm * N + gn] = val;
        } else if (STORE_MODE == 1) {
          ((bf16_t*)Cout)[cbase + (long)gm * N + gn] = (bf16_t)val;
        } else {
          const int bb = gm / Lrows;
          const int ll = gm - bb * Lrows;
          ((bf16_t*)Cout)[((long)bb * N + gn) * Lrows + ll] = (bf16_t)val;
        }
      }
    }
  }
}

// -------------------------------------------------------------- row softmax
__global__ __launch_bounds__(256) void softmax_rows(float* __restrict__ attn,
                                                    bf16_t* __restrict__ attn_bf,
                                                    const int* __restrict__ maskedPtr) {
  __shared__ float red[256];
  const int tid = threadIdx.x;
  const int row = blockIdx.x;
  const long base = ((long)blockIdx.y * MHA_L + row) * MHA_L;
  float* p = attn + base;
  bf16_t* pb = attn_bf + base;
  const int masked = maskedPtr[0];

  float x[8];
  float m = -INFINITY;
#pragma unroll
  for (int t = 0; t < 8; ++t) {
    const int j = tid + t * 256;
    float v = p[j];
    if (masked && j <= row) v = -INFINITY;  // reference fills where tril true
    x[t] = v;
    m = fmaxf(m, v);
  }
  red[tid] = m;
  __syncthreads();
  for (int s = 128; s > 0; s >>= 1) {
    if (tid < s) red[tid] = fmaxf(red[tid], red[tid + s]);
    __syncthreads();
  }
  m = red[0];
  __syncthreads();

  float sum = 0.f;
#pragma unroll
  for (int t = 0; t < 8; ++t) {
    const float e = __expf(x[t] - m);
    x[t] = e;
    sum += e;
  }
  red[tid] = sum;
  __syncthreads();
  for (int s = 128; s > 0; s >>= 1) {
    if (tid < s) red[tid] += red[tid + s];
    __syncthreads();
  }
  const float inv = 1.0f / red[0];
#pragma unroll
  for (int t = 0; t < 8; ++t) {
    const int j = tid + t * 256;
    const float r = x[t] * inv;
    p[j] = r;
    pb[j] = (bf16_t)r;
  }
}

// ------------------------------------------------------------------- driver
extern "C" void kernel_launch(void* const* d_in, const int* in_sizes, int n_in,
                              void* d_out, int out_size, void* d_ws, size_t ws_size,
                              hipStream_t stream) {
  (void)in_sizes; (void)n_in; (void)out_size; (void)ws_size;
  const float* Q  = (const float*)d_in[0];
  const float* Kx = (const float*)d_in[1];
  const float* V  = (const float*)d_in[2];
  const float* WQ = (const float*)d_in[3];
  const float* WK = (const float*)d_in[4];
  const float* WV = (const float*)d_in[5];
  const float* WO = (const float*)d_in[6];
  const int* maskedP = (const int*)d_in[7];

  const long BLD = (long)MHA_B * MHA_L * MHA_D;  // 8,388,608
  const long DD  = (long)MHA_D * MHA_D;          // 1,048,576

  float* out_f  = (float*)d_out;     // [B,L,D]
  float* attn_f = out_f + BLD;       // [B,L,L] (scores in place -> softmax)

  char* ws = (char*)d_ws;
  bf16_t* wq   = (bf16_t*)(ws);                      // 2 MB
  bf16_t* wk   = (bf16_t*)(ws + (2l << 20));         // 2 MB
  bf16_t* wv   = (bf16_t*)(ws + (4l << 20));         // 2 MB
  bf16_t* wo   = (bf16_t*)(ws + (6l << 20));         // 2 MB
  bf16_t* abuf = (bf16_t*)(ws + (8l << 20));         // 16 MB: Q/K/V bf16, later x
  bf16_t* qp   = (bf16_t*)(ws + (24l << 20));        // 16 MB
  bf16_t* kp   = (bf16_t*)(ws + (40l << 20));        // 16 MB
  bf16_t* vt   = (bf16_t*)(ws + (56l << 20));        // 16 MB (V^T per batch)
  bf16_t* attnb = qp;  // 32 MB alias over qp+kp (dead after score GEMM)

  const int M = MHA_B * MHA_L;  // 8192
  dim3 gProj(MHA_D / BN, M / BM, 1);             // (8, 64)
  dim3 gScores(MHA_L / BN, MHA_L / BM, MHA_B);   // (16, 16, 4)
  dim3 gAV(MHA_D / BN, MHA_L / BM, MHA_B);       // (8, 16, 4)

  cvt_f32_bf16<<<dim3(DD / 1024), 256, 0, stream>>>(WQ, wq, DD);
  cvt_f32_bf16<<<dim3(DD / 1024), 256, 0, stream>>>(WK, wk, DD);
  cvt_f32_bf16<<<dim3(DD / 1024), 256, 0, stream>>>(WV, wv, DD);
  cvt_f32_bf16<<<dim3(DD / 1024), 256, 0, stream>>>(WO, wo, DD);

  // q = Q @ WQ^T
  cvt_f32_bf16<<<dim3(BLD / 1024), 256, 0, stream>>>(Q, abuf, BLD);
  gemm_bf16_nt<1><<<gProj, 128, 0, stream>>>(abuf, wq, qp, M, MHA_D, MHA_D,
                                             0, 0, 0, 1.0f, MHA_L);
  // k = K @ WK^T
  cvt_f32_bf16<<<dim3(BLD / 1024), 256, 0, stream>>>(Kx, abuf, BLD);
  gemm_bf16_nt<1><<<gProj, 128, 0, stream>>>(abuf, wk, kp, M, MHA_D, MHA_D,
                                             0, 0, 0, 1.0f, MHA_L);
  // v = V @ WV^T, stored transposed per batch: vt[b][e][l]
  cvt_f32_bf16<<<dim3(BLD / 1024), 256, 0, stream>>>(V, abuf, BLD);
  gemm_bf16_nt<2><<<gProj, 128, 0, stream>>>(abuf, wv, vt, M, MHA_D, MHA_D,
                                             0, 0, 0, 1.0f, MHA_L);
  // scores = q @ k^T / 8 -> f32 directly into d_out attn region
  gemm_bf16_nt<0><<<gScores, 128, 0, stream>>>(
      qp, kp, attn_f, MHA_L, MHA_L, MHA_D,
      (long)MHA_L * MHA_D, (long)MHA_L * MHA_D, (long)MHA_L * MHA_L,
      0.125f, MHA_L);
  // softmax in place (+ bf16 copy for next GEMM)
  softmax_rows<<<dim3(MHA_L, MHA_B), 256, 0, stream>>>(attn_f, attnb, maskedP);
  // x = attn @ v  (NT against v^T), bf16 into abuf
  gemm_bf16_nt<1><<<gAV, 128, 0, stream>>>(
      attnb, vt, abuf, MHA_L, MHA_D, MHA_L,
      (long)MHA_L * MHA_L, (long)MHA_D * MHA_L, (long)MHA_L * MHA_D,
      1.0f, MHA_L);
  // out = x @ WO^T -> f32 d_out
  gemm_bf16_nt<0><<<gProj, 128, 0, stream>>>(abuf, wo, out_f, M, MHA_D, MHA_D,
                                             0, 0, 0, 1.0f, MHA_L);
}
